// CREDALoss_28638841930287
// MI455X (gfx1250) — matrix operand order repaired and checked by reading, not compiled, not run on hardware
//
#include <hip/hip_runtime.h>
#include <hip/hip_bf16.h>

typedef __bf16 bf16_t;
typedef __attribute__((ext_vector_type(16))) __bf16 v16bf;
typedef __attribute__((ext_vector_type(8)))  float  v8f;
typedef __attribute__((ext_vector_type(4)))  unsigned int uint4v;

#define NCLS 4

// ---------------------------------------------------------------------------
// helpers
// ---------------------------------------------------------------------------
union V16U { uint4v u[2]; v16bf v; };

// Load 16 bf16 as two 16-byte chunks at +0 and +off2 elements.
__device__ inline v16bf ld_pair(const bf16_t* p, int off2) {
  V16U t;
  t.u[0] = *(const uint4v*)(p);
  t.u[1] = *(const uint4v*)(p + off2);
  return t.v;
}

__device__ inline v8f wmma_bf16(v16bf a, v16bf b, v8f c) {
  // (neg_a, A, neg_b, B, c_mod, C, reuse_a, reuse_b)
  return __builtin_amdgcn_wmma_f32_16x16x32_bf16(false, a, false, b, (short)0, c,
                                                 false, false);
}

// ---------------------------------------------------------------------------
// 1) fp32 -> (hi, lo) bf16 split + row squared norms. One 256-thread block/row.
// ---------------------------------------------------------------------------
__global__ __launch_bounds__(256)
void convert_rows_kernel(const float* __restrict__ src,
                         bf16_t* __restrict__ hi, bf16_t* __restrict__ lo,
                         float* __restrict__ norms, int D) {
  const int row = blockIdx.x;
  const size_t base = (size_t)row * D;
  float s = 0.f;
  for (int c = threadIdx.x; c < D; c += 256) {
    float x = src[base + c];
    s += x * x;
    bf16_t h = (bf16_t)x;
    bf16_t l = (bf16_t)(x - (float)h);
    hi[base + c] = h;
    lo[base + c] = l;
  }
  __shared__ float red[256];
  red[threadIdx.x] = s;
  __syncthreads();
  for (int off = 128; off > 0; off >>= 1) {
    if (threadIdx.x < off) red[threadIdx.x] += red[threadIdx.x + off];
    __syncthreads();
  }
  if (threadIdx.x == 0) norms[row] = red[0];
}

// ---------------------------------------------------------------------------
// 2a) source rows: cross-entropy sum + class counts
//     accum[0] += sum(-logp[label]);  accum[2+c] += count
// ---------------------------------------------------------------------------
__global__ __launch_bounds__(256)
void src_stats_kernel(const float* __restrict__ logits,
                      const int* __restrict__ labels,
                      float* __restrict__ accum, int n) {
  __shared__ float sCls;
  __shared__ float sCnt[NCLS];
  if (threadIdx.x == 0) sCls = 0.f;
  if (threadIdx.x < NCLS) sCnt[threadIdx.x] = 0.f;
  __syncthreads();
  int i = blockIdx.x * 256 + threadIdx.x;
  if (i < n) {
    float x0 = logits[i*4+0], x1 = logits[i*4+1], x2 = logits[i*4+2], x3 = logits[i*4+3];
    float m = fmaxf(fmaxf(x0, x1), fmaxf(x2, x3));
    float se = __expf(x0 - m) + __expf(x1 - m) + __expf(x2 - m) + __expf(x3 - m);
    float lse = m + logf(se);
    int lab = labels[i];
    float xl = logits[i*4 + lab];
    atomicAdd(&sCls, lse - xl);           // -log p[label]
    atomicAdd(&sCnt[lab], 1.0f);
  }
  __syncthreads();
  if (threadIdx.x == 0) atomicAdd(&accum[0], sCls);
  if (threadIdx.x < NCLS) atomicAdd(&accum[2 + threadIdx.x], sCnt[threadIdx.x]);
}

// ---------------------------------------------------------------------------
// 2b) target rows: entropy sum, pseudo labels, uncertainty weights w^2
//     accum[1] += entropy;  accum[6+c] += count;  accum[10+c] += w^2
// ---------------------------------------------------------------------------
__global__ __launch_bounds__(256)
void tgt_stats_kernel(const float* __restrict__ logits,
                      float* __restrict__ accum,
                      int* __restrict__ pseudo, float* __restrict__ wt2arr, int n) {
  __shared__ float sEnt;
  __shared__ float sNt[NCLS];
  __shared__ float sTr[NCLS];
  if (threadIdx.x == 0) sEnt = 0.f;
  if (threadIdx.x < NCLS) { sNt[threadIdx.x] = 0.f; sTr[threadIdx.x] = 0.f; }
  __syncthreads();
  int i = blockIdx.x * 256 + threadIdx.x;
  if (i < n) {
    const float EPS = 1e-8f;
    const float INVLOG2 = 1.4426950408889634f;  // 1/ln(2)
    float x[4] = {logits[i*4+0], logits[i*4+1], logits[i*4+2], logits[i*4+3]};
    float m = fmaxf(fmaxf(x[0], x[1]), fmaxf(x[2], x[3]));
    float e[4], se = 0.f;
    for (int c = 0; c < 4; ++c) { e[c] = __expf(x[c] - m); se += e[c]; }
    float inv = 1.0f / se;
    float ent = 0.f, p2 = 0.f;
    float best = -1.f; int am = 0;
    for (int c = 0; c < 4; ++c) {
      float p = e[c] * inv;
      ent -= p * logf(p + EPS);
      p2 += p * p;
      if (p > best) { best = p; am = c; }
    }
    float h2p = -logf(p2 + EPS) * INVLOG2;
    float h2max = 2.0f;                    // log2(4)
    float w = 1.0f - h2p / (h2max + EPS);
    float w2 = w * w;
    pseudo[i] = am;
    wt2arr[i] = w2;
    atomicAdd(&sEnt, ent);
    atomicAdd(&sNt[am], 1.0f);
    atomicAdd(&sTr[am], w2);
  }
  __syncthreads();
  if (threadIdx.x == 0) atomicAdd(&accum[1], sEnt);
  if (threadIdx.x < NCLS) {
    atomicAdd(&accum[6 + threadIdx.x],  sNt[threadIdx.x]);
    atomicAdd(&accum[10 + threadIdx.x], sTr[threadIdx.x]);
  }
}

// ---------------------------------------------------------------------------
// 3) Fused GEMM + exp + masked per-class reduction.
//    Block = 256 threads = 8 waves; wave tile = 32x32 (4 WMMA C tiles).
//    Block tile = 64 rows x 128 cols.  dot = Ah*Bh + Ah*Bl + Al*Bh (split bf16).
// ---------------------------------------------------------------------------
__device__ inline void tile_epilogue(v8f c, int mBase, int nBase, int lane,
                                     const float* __restrict__ normA,
                                     const float* __restrict__ normB,
                                     const int* __restrict__ clsA,
                                     const int* __restrict__ clsB,
                                     const float* __restrict__ wA,
                                     const float* __restrict__ wB,
                                     float kscale, float acc[NCLS]) {
  // C/D layout: VGPR r, lanes 0-15 -> M=r, N=lane; lanes 16-31 -> M=r+8, N=lane-16
  const int n  = nBase + (lane & 15);
  const int cc = clsB[n];
  const float cw = wB ? wB[n] : 1.0f;
  const float nb = normB[n];
  const int mOff = mBase + ((lane >> 4) << 3);
#pragma unroll
  for (int r = 0; r < 8; ++r) {
    int m = mOff + r;
    float s  = c[r];
    float d2 = fmaxf(normA[m] + nb - 2.0f * s, 0.0f);
    float k2 = __expf(-d2 * kscale);             // K^2 = exp(-2 d2/(2*sigma^2+eps))
    int rc = clsA[m];
    float rw = wA ? wA[m] : 1.0f;
    float v = (rc == cc) ? rw * cw * k2 : 0.0f;
#pragma unroll
    for (int cidx = 0; cidx < NCLS; ++cidx)
      acc[cidx] += (rc == cidx) ? v : 0.0f;
  }
}

__global__ __launch_bounds__(256)
void rbf_class_sums_kernel(const bf16_t* __restrict__ Ahi, const bf16_t* __restrict__ Alo,
                           const float* __restrict__ normA,
                           const bf16_t* __restrict__ Bhi, const bf16_t* __restrict__ Blo,
                           const float* __restrict__ normB,
                           const int* __restrict__ clsA, const int* __restrict__ clsB,
                           const float* __restrict__ wA, const float* __restrict__ wB,
                           float* __restrict__ out4, int D, float kscale) {
  const int tid  = threadIdx.x;
  const int lane = tid & 31;
  const int wave = tid >> 5;
  const int wm = wave & 1;
  const int wn = wave >> 1;
  const int rowBase = blockIdx.y * 64  + wm * 32;
  const int colBase = blockIdx.x * 128 + wn * 32;

  const int lane16 = lane & 15;
  const int half   = lane >> 4;   // 0 or 1

  // A 16x32 bf16 layout: lanes 0-15 -> K 0..15 (contiguous), lanes 16-31 -> K 16..31
  const size_t aOff0 = (size_t)(rowBase + lane16) * D + half * 16;
  const size_t aOff1 = aOff0 + (size_t)16 * D;
  // B 32x16 bf16 layout: lanes 0-15 -> K 0..15 (contiguous), lanes 16-31 -> K 16..31
  const size_t bOff0 = (size_t)(colBase + lane16) * D + half * 16;
  const size_t bOff1 = bOff0 + (size_t)16 * D;

  v8f c00 = {}; v8f c01 = {}; v8f c10 = {}; v8f c11 = {};

  for (int kk = 0; kk < D; kk += 32) {
    v16bf a0h = ld_pair(Ahi + aOff0 + kk, 8);
    v16bf a0l = ld_pair(Alo + aOff0 + kk, 8);
    v16bf a1h = ld_pair(Ahi + aOff1 + kk, 8);
    v16bf a1l = ld_pair(Alo + aOff1 + kk, 8);
    v16bf b0h = ld_pair(Bhi + bOff0 + kk, 8);
    v16bf b0l = ld_pair(Blo + bOff0 + kk, 8);
    v16bf b1h = ld_pair(Bhi + bOff1 + kk, 8);
    v16bf b1l = ld_pair(Blo + bOff1 + kk, 8);

    c00 = wmma_bf16(a0h, b0h, c00);
    c01 = wmma_bf16(a0h, b1h, c01);
    c10 = wmma_bf16(a1h, b0h, c10);
    c11 = wmma_bf16(a1h, b1h, c11);
    c00 = wmma_bf16(a0h, b0l, c00);
    c01 = wmma_bf16(a0h, b1l, c01);
    c10 = wmma_bf16(a1h, b0l, c10);
    c11 = wmma_bf16(a1h, b1l, c11);
    c00 = wmma_bf16(a0l, b0h, c00);
    c01 = wmma_bf16(a0l, b1h, c01);
    c10 = wmma_bf16(a1l, b0h, c10);
    c11 = wmma_bf16(a1l, b1h, c11);
  }

  float acc[NCLS] = {0.f, 0.f, 0.f, 0.f};
  tile_epilogue(c00, rowBase,      colBase,      lane, normA, normB, clsA, clsB, wA, wB, kscale, acc);
  tile_epilogue(c01, rowBase,      colBase + 16, lane, normA, normB, clsA, clsB, wA, wB, kscale, acc);
  tile_epilogue(c10, rowBase + 16, colBase,      lane, normA, normB, clsA, clsB, wA, wB, kscale, acc);
  tile_epilogue(c11, rowBase + 16, colBase + 16, lane, normA, normB, clsA, clsB, wA, wB, kscale, acc);

  __shared__ float sacc[NCLS];
  if (tid < NCLS) sacc[tid] = 0.f;
  __syncthreads();
#pragma unroll
  for (int c = 0; c < NCLS; ++c) atomicAdd(&sacc[c], acc[c]);
  __syncthreads();
  if (tid < NCLS) atomicAdd(&out4[tid], sacc[tid]);
}

// ---------------------------------------------------------------------------
// 4) finalize: H2 algebra + total loss
// ---------------------------------------------------------------------------
__global__ void zero_accum_kernel(float* a) {
  if (threadIdx.x < 32) a[threadIdx.x] = 0.f;
}

__global__ void finalize_kernel(const float* __restrict__ accum,
                                float* __restrict__ out, int nS, int nT) {
  if (threadIdx.x != 0 || blockIdx.x != 0) return;
  const float EPS = 1e-8f;
  const float INVLOG2 = 1.4426950408889634f;
  float loss_cls = accum[0] / (float)nS;
  float loss_ent = accum[1] / (float)nT;
  float creda_sum = 0.f, n_valid = 0.f;
  for (int c = 0; c < NCLS; ++c) {
    float ns   = accum[2 + c];
    float nt   = accum[6 + c];
    float tr_t = accum[10 + c];
    float ss_s  = accum[16 + c];
    float ss_t  = accum[20 + c];
    float ss_st = accum[24 + c];
    float tr_s = ns;
    float ds = tr_s + EPS;
    float dt = tr_t + EPS;
    float dm = tr_s + tr_t + EPS;
    float h_s = -logf(ss_s / (ds * ds) + EPS) * INVLOG2;
    float h_t = -logf(ss_t / (dt * dt) + EPS) * INVLOG2;
    float h_m = -logf((ss_s + 2.f * ss_st + ss_t) / (dm * dm) + EPS) * INVLOG2;
    float pc = h_m - 0.5f * (h_s + h_t);
    if (ns >= 2.f && nt >= 2.f) { creda_sum += pc; n_valid += 1.f; }
  }
  float loss_creda = (n_valid > 0.f) ? creda_sum / fmaxf(n_valid, 1.f) : 0.f;
  out[0] = loss_cls + 1.0f * loss_creda + 0.1f * loss_ent;
}

// ---------------------------------------------------------------------------
// launch
// ---------------------------------------------------------------------------
extern "C" void kernel_launch(void* const* d_in, const int* in_sizes, int n_in,
                              void* d_out, int out_size, void* d_ws, size_t ws_size,
                              hipStream_t stream) {
  const float* feat_s = (const float*)d_in[0];
  const float* log_s  = (const float*)d_in[1];
  const float* feat_t = (const float*)d_in[2];
  const float* log_t  = (const float*)d_in[3];
  const int*   lab_s  = (const int*)d_in[4];

  const int NS = in_sizes[4];
  const int D  = in_sizes[0] / NS;
  const int NT = in_sizes[2] / D;

  // workspace carve-out (256B aligned)
  char* w = (char*)d_ws;
  size_t off = 0;
  auto alloc = [&](size_t bytes) -> char* {
    char* p = w + off;
    off += (bytes + 255) & ~(size_t)255;
    return p;
  };
  bf16_t* s_hi   = (bf16_t*)alloc((size_t)NS * D * sizeof(bf16_t));
  bf16_t* s_lo   = (bf16_t*)alloc((size_t)NS * D * sizeof(bf16_t));
  bf16_t* t_hi   = (bf16_t*)alloc((size_t)NT * D * sizeof(bf16_t));
  bf16_t* t_lo   = (bf16_t*)alloc((size_t)NT * D * sizeof(bf16_t));
  float*  norm_s = (float*)alloc((size_t)NS * sizeof(float));
  float*  norm_t = (float*)alloc((size_t)NT * sizeof(float));
  float*  wt2    = (float*)alloc((size_t)NT * sizeof(float));
  int*    pseudo = (int*)alloc((size_t)NT * sizeof(int));
  float*  accum  = (float*)alloc(32 * sizeof(float));

  zero_accum_kernel<<<1, 32, 0, stream>>>(accum);
  convert_rows_kernel<<<NS, 256, 0, stream>>>(feat_s, s_hi, s_lo, norm_s, D);
  convert_rows_kernel<<<NT, 256, 0, stream>>>(feat_t, t_hi, t_lo, norm_t, D);
  src_stats_kernel<<<(NS + 255) / 256, 256, 0, stream>>>(log_s, lab_s, accum, NS);
  tgt_stats_kernel<<<(NT + 255) / 256, 256, 0, stream>>>(log_t, accum, pseudo, wt2, NT);

  // K^2 = exp(-2*d2 / (2*sigma^2 + eps)), sigma = 32
  const float kscale = 2.0f / (2.0f * 32.0f * 32.0f + 1e-8f);
  dim3 blk(256);

  // ss_s: source x source, label mask, unit weights
  rbf_class_sums_kernel<<<dim3(NS / 128, NS / 64), blk, 0, stream>>>(
      s_hi, s_lo, norm_s, s_hi, s_lo, norm_s, lab_s, lab_s,
      nullptr, nullptr, accum + 16, D, kscale);
  // ss_t: target x target, pseudo mask, w^2 weights both sides
  rbf_class_sums_kernel<<<dim3(NT / 128, NT / 64), blk, 0, stream>>>(
      t_hi, t_lo, norm_t, t_hi, t_lo, norm_t, pseudo, pseudo,
      wt2, wt2, accum + 20, D, kscale);
  // ss_st: source rows x target cols, label/pseudo masks, unit weights
  rbf_class_sums_kernel<<<dim3(NT / 128, NS / 64), blk, 0, stream>>>(
      s_hi, s_lo, norm_s, t_hi, t_lo, norm_t, lab_s, pseudo,
      nullptr, nullptr, accum + 24, D, kscale);

  finalize_kernel<<<1, 1, 0, stream>>>(accum, (float*)d_out, NS, NT);
}
